// GraphAttentionLayer_7859790152023
// MI455X (gfx1250) — matrix operand order repaired and checked by reading, not compiled, not run on hardware
//
#include <hip/hip_runtime.h>

typedef __attribute__((ext_vector_type(16))) _Float16 v16h;
typedef __attribute__((ext_vector_type(8)))  _Float16 v8h;
typedef __attribute__((ext_vector_type(8)))  float    v8f;
typedef __attribute__((ext_vector_type(4)))  float    v4f;
typedef __attribute__((ext_vector_type(4)))  int      v4i;
typedef __attribute__((ext_vector_type(4)))  unsigned v4u;
typedef __attribute__((ext_vector_type(8)))  int      v8i;

#define NTOT  4096
#define FIN   256
#define FOUT  128
#define BATCH 4

// ---------------------------------------------------------------------------
// Kernel 0: convert W (FOUT x FIN, row-major fp32) to f16 (same layout).
// ---------------------------------------------------------------------------
__global__ void k_convert_w(const float* __restrict__ W,
                            _Float16* __restrict__ W16, int n) {
    int t = blockIdx.x * 256 + threadIdx.x;
    if (t < n) W16[t] = (_Float16)W[t];
}

// ---------------------------------------------------------------------------
// Kernel 1: Wh = h @ W^T + b via v_wmma_f32_16x16x32_f16.
// One wave -> 16-row x 128-col output tile. Output stored TRANSPOSED as f16:
// WhT[b][o][n] (o-major) so kernel 3 loads B-fragments with contiguous reads.
// ---------------------------------------------------------------------------
__global__ void k_gemm_wh(const float* __restrict__ h,
                          const _Float16* __restrict__ W16,
                          const float* __restrict__ Wb,
                          _Float16* __restrict__ WhT) {
    const int lane = threadIdx.x & 31;
    const int wave = threadIdx.x >> 5;          // 0..3
    const int r    = lane & 15;
    const int hf   = lane >> 4;                 // half-wave id
    const int mtile = blockIdx.x * 4 + wave;    // 0..1023
    const int mbase = mtile * 16;
    const int b     = mbase >> 12;
    const int nloc  = mbase & 4095;

    v8f zero = {};
    v8f acc[8];
#pragma unroll
    for (int ob = 0; ob < 8; ++ob) acc[ob] = zero;

    const float* hrow = h + (size_t)(mbase + r) * FIN;

    for (int k0 = 0; k0 < FIN; k0 += 32) {
        // A fragment: lane (r,hf) holds h[mbase+r][k0 + {hf*8+0..7, 16+hf*8+0..7}]
        v16h afrag;
        {
            const float* p0 = hrow + k0 + hf * 8;
            v4f x0 = *(const v4f*)(p0);
            v4f x1 = *(const v4f*)(p0 + 4);
            v4f x2 = *(const v4f*)(p0 + 16);
            v4f x3 = *(const v4f*)(p0 + 20);
#pragma unroll
            for (int i = 0; i < 4; ++i) {
                afrag[i]      = (_Float16)x0[i];
                afrag[4 + i]  = (_Float16)x1[i];
                afrag[8 + i]  = (_Float16)x2[i];
                afrag[12 + i] = (_Float16)x3[i];
            }
        }
#pragma unroll
        for (int ob = 0; ob < 8; ++ob) {
            // B fragment: column o = ob*16 + r, K = k0 + {hf*8+0..7, 16+hf*8+0..7}
            const _Float16* wp = W16 + (size_t)(ob * 16 + r) * FIN + k0 + hf * 8;
            v8h b0 = *(const v8h*)(wp);
            v8h b1 = *(const v8h*)(wp + 16);
            v16h bfrag;
#pragma unroll
            for (int i = 0; i < 8; ++i) { bfrag[i] = b0[i]; bfrag[8 + i] = b1[i]; }
            acc[ob] = __builtin_amdgcn_wmma_f32_16x16x32_f16(
                false, afrag, false, bfrag, (short)0, acc[ob], false, false);
        }
    }

    // Store transposed (+bias): D lane holds (m = v + 8*hf, o = ob*16 + r)
#pragma unroll
    for (int ob = 0; ob < 8; ++ob) {
        const int o = ob * 16 + r;
        const float bias = Wb[o];
        v8h outv;
#pragma unroll
        for (int v = 0; v < 8; ++v) outv[v] = (_Float16)(acc[ob][v] + bias);
        _Float16* dst = WhT + (size_t)(b * FOUT + o) * NTOT + nloc + hf * 8;
        *(v8h*)dst = outv;
    }
}

// ---------------------------------------------------------------------------
// Kernel 2: src[b][n] = sum_o WhT[b][o][n]*a_src[o] + a_b ; dst analogous.
// ---------------------------------------------------------------------------
__global__ void k_srcdst(const _Float16* __restrict__ WhT,
                         const float* __restrict__ a_w,
                         const float* __restrict__ a_b,
                         float* __restrict__ srcb,
                         float* __restrict__ dstb) {
    int t = blockIdx.x * 256 + threadIdx.x;     // 16384 threads
    int b = t >> 12, n = t & 4095;
    const _Float16* base = WhT + (size_t)b * FOUT * NTOT + n;
    float s = 0.f, d = 0.f;
#pragma unroll 4
    for (int o = 0; o < FOUT; ++o) {
        float w = (float)base[(size_t)o * NTOT];
        s += w * a_w[o];
        d += w * a_w[FOUT + o];
    }
    srcb[t] = s + a_b[0];
    dstb[t] = d;
}

// ---------------------------------------------------------------------------
// TDM descriptor: load a 2D tile (tile_dim0=32 cols f16, tile_dim1=128 rows,
// row stride 4096 elements) from global WhT into LDS at lds_off.
//   Group0: count=1 | lds_addr | global_addr(57b) | type=2
//   Group1: data_size=1(2B) | tensor_dim0=4096 | tensor_dim1=128 |
//           tile_dim0=32 | tile_dim1=128 | tensor_dim0_stride=4096
//   Groups 2/3 (+extra group on this toolchain): zero (2-D tensor)
// ---------------------------------------------------------------------------
__device__ __forceinline__ void tdm_load_tile(const _Float16* gbase,
                                              unsigned lds_off) {
    unsigned long long ga = (unsigned long long)(const void*)gbase;
    v4u g0;
    g0[0] = 1u;                                   // count=1, user mode
    g0[1] = lds_off;                              // lds_addr (bytes)
    g0[2] = (unsigned)ga;                         // global_addr[31:0]
    g0[3] = (unsigned)(ga >> 32) | (2u << 30);    // global_addr[56:32] | type=2
    v8i g1;
    g1[0] = (int)(1u << 16);                      // data_size = 1 -> 2 bytes
    g1[1] = (int)(4096u << 16);                   // tensor_dim0 = 4096 (lo16)
    g1[2] = (int)(128u << 16);                    // tensor_dim1 = 128 (lo16)
    g1[3] = (int)(32u << 16);                     // tile_dim0 = 32
    g1[4] = 128;                                  // tile_dim1 = 128 (tile_dim2=0)
    g1[5] = 4096;                                 // tensor_dim0_stride (lo32)
    g1[6] = 0;
    g1[7] = 0;
    v4i gz = {0, 0, 0, 0};
    v8i gz8 = {0, 0, 0, 0, 0, 0, 0, 0};
    // 6-arg form on this toolchain: (g0, g1, g2, g3, g_extra, cpol)
    __builtin_amdgcn_tensor_load_to_lds(g0, g1, gz, gz, gz8, 0);
}

// ---------------------------------------------------------------------------
// Kernel 3: fused masked-softmax-attention: h' = softmax(mask(leaky(e))) @ Wh.
// One wave owns a 16-row i-tile (16x128 f32 accumulator = 8 o-blocks).
// Per 32-column chunk:
//   - wave 0 issues a TDM tensor_load_to_lds for chunk j+1 (double buffered),
//     waits TENSORcnt for chunk j; one workgroup barrier publishes the tile
//   - each lane computes 16 unnormalized exp() of its own row -> P A-fragment
//   - 8x v_wmma_f32_16x16x32_f16 accumulate P @ Wh from the LDS tile
// Normalize by row sums at the end (cross-lane combine via shfl).
// Grid: (64, 4) blocks x 128 threads (4 waves, 64 rows per block).
// ---------------------------------------------------------------------------
__global__ void k_attn(const int* __restrict__ adj,
                       const _Float16* __restrict__ WhT,
                       const float* __restrict__ srcb,
                       const float* __restrict__ dstb,
                       float* __restrict__ out) {
    __shared__ __align__(32) _Float16 tile[2][FOUT * 32];   // 2 x 8 KB

    const int lane  = threadIdx.x & 31;
    const int wave  = threadIdx.x >> 5;                  // 0..3
    const int r     = lane & 15;
    const int hf    = lane >> 4;
    const int b     = blockIdx.y;
    const int ibase = blockIdx.x * 64 + wave * 16;
    const int irow  = ibase + r;

    const float s_i = srcb[b * NTOT + irow];
    const int*      adjrow = adj  + ((size_t)b * NTOT + irow) * NTOT;
    const float*    dstrow = dstb + b * NTOT;
    const _Float16* whb    = WhT  + (size_t)b * FOUT * NTOT;

    // LDS byte offsets of the two tile buffers (low 32 bits of flat address).
    const unsigned lds0 = (unsigned)(size_t)(const void*)&tile[0][0];
    const unsigned lds1 = (unsigned)(size_t)(const void*)&tile[1][0];

    v8f zero = {};
    v8f acc[8];
#pragma unroll
    for (int ob = 0; ob < 8; ++ob) acc[ob] = zero;
    float rowsum = 0.f;

    // Prologue: DMA chunk 0 into buffer 0.
    if (wave == 0) tdm_load_tile(whb, lds0);

    for (int j0 = 0; j0 < NTOT; j0 += 32) {
        const int cur = (j0 >> 5) & 1;
        if (wave == 0) __builtin_amdgcn_s_wait_tensorcnt(0);  // chunk j0 landed
        __syncthreads();   // publish tile[cur]; all waves done with tile[cur^1]
        if (wave == 0 && (j0 + 32) < NTOT)
            tdm_load_tile(whb + j0 + 32, cur ? lds0 : lds1);

        const _Float16* tbuf = &tile[cur][0];

        // Prefetch the adj cachelines of the next chunk (streaming read).
        if (j0 + 32 < NTOT) __builtin_prefetch(adjrow + j0 + 32, 0, 1);

        // P fragment: lane handles row r, 16 j's: {hf*8+0..7, 16+hf*8+0..7}
        v16h pfrag;
        {
            const float* dp = dstrow + j0 + hf * 8;
            v4f d0 = *(const v4f*)(dp);
            v4f d1 = *(const v4f*)(dp + 4);
            v4f d2 = *(const v4f*)(dp + 16);
            v4f d3 = *(const v4f*)(dp + 20);
            const int* ap = adjrow + j0 + hf * 8;
            v4i a0 = *(const v4i*)(ap);
            v4i a1 = *(const v4i*)(ap + 4);
            v4i a2 = *(const v4i*)(ap + 16);
            v4i a3 = *(const v4i*)(ap + 20);
            float dv[16]; int av[16];
#pragma unroll
            for (int i = 0; i < 4; ++i) {
                dv[i] = d0[i]; dv[4+i] = d1[i]; dv[8+i] = d2[i]; dv[12+i] = d3[i];
                av[i] = a0[i]; av[4+i] = a1[i]; av[8+i] = a2[i]; av[12+i] = a3[i];
            }
#pragma unroll
            for (int i = 0; i < 16; ++i) {
                float e = s_i + dv[i];
                e = (e >= 0.f) ? e : 0.01f * e;             // leaky relu
                float p = 0.f;
                if (av[i] > 0) p = __expf(fminf(e, 11.0f)); // mask; f16-safe
                rowsum += p;
                pfrag[i] = (_Float16)p;
            }
        }
#pragma unroll
        for (int ob = 0; ob < 8; ++ob) {
            // B fragment: column o = ob*16 + r, K = {hf*8+0..7, 16+hf*8+0..7} of this chunk
            const _Float16* tp = tbuf + (ob * 16 + r) * 32 + hf * 8;
            v8h b0 = *(const v8h*)(tp);
            v8h b1 = *(const v8h*)(tp + 16);
            v16h bfrag;
#pragma unroll
            for (int i = 0; i < 8; ++i) { bfrag[i] = b0[i]; bfrag[8 + i] = b1[i]; }
            acc[ob] = __builtin_amdgcn_wmma_f32_16x16x32_f16(
                false, pfrag, false, bfrag, (short)0, acc[ob], false, false);
        }
    }

    // Row sum lives split across lane pair (l, l^16): combine, invert.
    float total = rowsum + __shfl_xor(rowsum, 16, 32);
    float inv = (total > 0.f) ? (1.0f / total) : 0.f;
    // Accumulator lane owns rows m = v + 8*hf; inv for row m sits in lane m.
    float invs[8];
#pragma unroll
    for (int v = 0; v < 8; ++v) invs[v] = __shfl(inv, v + hf * 8, 32);

    float* orow = out + ((size_t)b * NTOT + ibase) * FOUT;
#pragma unroll
    for (int ob = 0; ob < 8; ++ob) {
        const int o = ob * 16 + r;
#pragma unroll
        for (int v = 0; v < 8; ++v) {
            orow[(size_t)(v + 8 * hf) * FOUT + o] = acc[ob][v] * invs[v];
        }
    }
}

// ---------------------------------------------------------------------------
// Launch. Workspace layout (needs ~4.4 MB):
//   [0,       64KB)  W16   (128x256 f16)
//   [64KB,   4.25MB) WhT   (4x128x4096 f16)
//   [+0,      64KB)  srcb  (4x4096 f32)
//   [+64KB,  128KB)  dstb  (4x4096 f32)
// ---------------------------------------------------------------------------
extern "C" void kernel_launch(void* const* d_in, const int* in_sizes, int n_in,
                              void* d_out, int out_size, void* d_ws, size_t ws_size,
                              hipStream_t stream) {
    const float* h    = (const float*)d_in[0];
    const int*   adj  = (const int*)d_in[1];
    const float* W_w  = (const float*)d_in[2];
    const float* W_b  = (const float*)d_in[3];
    const float* a_w  = (const float*)d_in[4];
    const float* a_b  = (const float*)d_in[5];
    float* out = (float*)d_out;

    char* ws = (char*)d_ws;
    _Float16* W16 = (_Float16*)ws;
    _Float16* WhT = (_Float16*)(ws + 65536);
    float* srcb = (float*)(ws + 65536 + (size_t)BATCH * FOUT * NTOT * 2);
    float* dstb = srcb + BATCH * NTOT;

    k_convert_w<<<128, 256, 0, stream>>>(W_w, W16, FOUT * FIN);
    k_gemm_wh<<<256, 128, 0, stream>>>(h, W16, W_b, WhT);
    k_srcdst<<<64, 256, 0, stream>>>(WhT, a_w, a_b, srcb, dstb);
    dim3 g(NTOT / 64, BATCH);
    k_attn<<<g, 128, 0, stream>>>(adj, WhT, srcb, dstb, out);
}
